// SlotAttention_85461259256217
// MI455X (gfx1250) — compile-verified, matrix operand order checked by reading, not run on hardware
//
#include <hip/hip_runtime.h>
#include <hip/hip_bf16.h>
#include <math.h>

typedef float v2f  __attribute__((ext_vector_type(2)));
typedef float v4f  __attribute__((ext_vector_type(4)));
typedef float v8f  __attribute__((ext_vector_type(8)));
typedef __bf16 v8bf  __attribute__((ext_vector_type(8)));
typedef __bf16 v16bf __attribute__((ext_vector_type(16)));

#define NUM_ITER_ 3
#define S_SLOTS 8
#define D_DIM 128
#define H_DIM 256
#define B_BATCH 128
#define N_TOK 4096
#define GROUPB 32            // batches per L2-resident group (k+v bf16 = 64 MB < 192 MB L2)
#define LN_EPS_ 1e-5f
#define EPS_ 1e-8f

static __device__ __forceinline__ v8f wmma_f32(v2f a, v2f b, v8f c) {
  // D(16x16,f32) = A(16x4,f32) * B(4x16,f32) + C
  return __builtin_amdgcn_wmma_f32_16x16x4_f32(false, a, false, b, (short)0, c, false, false);
}
static __device__ __forceinline__ v16bf cat8(v8bf a, v8bf b) {
  return __builtin_shufflevector(a, b, 0,1,2,3,4,5,6,7,8,9,10,11,12,13,14,15);
}
static __device__ __forceinline__ v8f wmma_bf16(v16bf a, v16bf b, v8f c) {
  // D(16x16,f32) = A(16x32,bf16) * B(32x16,bf16) + C
  return __builtin_amdgcn_wmma_f32_16x16x32_bf16(false, a, false, b, (short)0, c, false, false);
}

// ---------------------------------------------------------------------------
// slots = mu + exp(log_sigma) * noise   (whole batch, once)
// ---------------------------------------------------------------------------
__global__ void kn_init(const float* __restrict__ noise, const float* __restrict__ mu,
                        const float* __restrict__ ls, float* __restrict__ slots) {
  int i = blockIdx.x * 256 + threadIdx.x;
  if (i < B_BATCH * S_SLOTS * D_DIM) {
    int d = i & 127;
    slots[i] = mu[d] + __expf(ls[d]) * noise[i];
  }
}

// ---------------------------------------------------------------------------
// Fused LayerNorm(x) + k = xn Wk^T (bf16, row-major) + v = xn Wv^T (bf16,
// TRANSPOSED [d][n] for fragment-friendly reads).  f32 WMMA (16x16x4).
// x is streamed with non-temporal loads (read exactly once, staged in LDS)
// so it does not evict the L2-resident k/v group buffers.
// ---------------------------------------------------------------------------
__global__ void __launch_bounds__(256) kn_lnkv(const float* __restrict__ x,
                                               const float* __restrict__ Wk,
                                               const float* __restrict__ Wv,
                                               __bf16* __restrict__ kout,   // [GROUPB*N][128]
                                               __bf16* __restrict__ vtout) {// [GROUPB*128][N]
  __shared__ float s_x[32][132];
  __shared__ float s_mu[32];
  __shared__ float s_rs[32];
  const int t = threadIdx.x;
  const int w = t >> 5, L = t & 31;
  const size_t row0 = (size_t)blockIdx.x * 32;   // row within group (never crosses a batch)

  // --- stream x once (NT), stage raw in LDS, compute LN stats (8 lanes/row) ---
  {
    const int r = t >> 3;
    const int c0 = (t & 7) * 16;
    const float* xr = x + (row0 + r) * D_DIM + c0;
    float s = 0.f, s2 = 0.f;
    for (int i = 0; i < 4; ++i) {
      v4f vv = __builtin_nontemporal_load((const v4f*)(xr + 4 * i));
      s_x[r][c0 + 4 * i + 0] = vv.x; s_x[r][c0 + 4 * i + 1] = vv.y;
      s_x[r][c0 + 4 * i + 2] = vv.z; s_x[r][c0 + 4 * i + 3] = vv.w;
      s  += vv.x + vv.y + vv.z + vv.w;
      s2 += vv.x * vv.x + vv.y * vv.y + vv.z * vv.z + vv.w * vv.w;
    }
    s  += __shfl_xor(s, 1);  s  += __shfl_xor(s, 2);  s  += __shfl_xor(s, 4);
    s2 += __shfl_xor(s2, 1); s2 += __shfl_xor(s2, 2); s2 += __shfl_xor(s2, 4);
    if ((L & 7) == 0) {
      float mu = s * (1.f / 128.f);
      float var = s2 * (1.f / 128.f) - mu * mu;
      s_mu[r] = mu;
      s_rs[r] = rsqrtf(var + LN_EPS_);
    }
  }
  __syncthreads();

  // --- WMMA: wave w -> row-tile (w&1), col tiles {2*(w>>1), +1}, both k & v ---
  const int rt = w & 1;
  const int cb = (w >> 1) * 2;
  const int dl = (L < 16) ? 0 : 2;   // f32 frag K offset per lane half
  const int ml = L & 15;
  const int lrow = rt * 16 + ml;
  const float mu = s_mu[lrow], rs = s_rs[lrow];

  v8f kacc0 = {0,0,0,0,0,0,0,0}, kacc1 = {0,0,0,0,0,0,0,0};
  v8f vacc0 = {0,0,0,0,0,0,0,0}, vacc1 = {0,0,0,0,0,0,0,0};
  const int d0 = cb * 16 + ml, d1 = d0 + 16;
  const float* wk0 = Wk + (size_t)d0 * D_DIM;
  const float* wk1 = Wk + (size_t)d1 * D_DIM;
  const float* wv0 = Wv + (size_t)d0 * D_DIM;
  const float* wv1 = Wv + (size_t)d1 * D_DIM;

  for (int k0 = 0; k0 < D_DIM; k0 += 4) {
    v2f a = *(const v2f*)(&s_x[lrow][k0 + dl]);
    a.x = (a.x - mu) * rs;
    a.y = (a.y - mu) * rs;
    v2f b;
    b = *(const v2f*)(wk0 + k0 + dl); kacc0 = wmma_f32(a, b, kacc0);
    b = *(const v2f*)(wk1 + k0 + dl); kacc1 = wmma_f32(a, b, kacc1);
    b = *(const v2f*)(wv0 + k0 + dl); vacc0 = wmma_f32(a, b, vacc0);
    b = *(const v2f*)(wv1 + k0 + dl); vacc1 = wmma_f32(a, b, vacc1);
  }

  const int ro = (L < 16) ? 0 : 8;       // C/D frag: VGPR j holds rows j, j+8
  const size_t bl = row0 >> 12;          // local batch in group
  const size_t nb = row0 & (N_TOK - 1);  // n offset within batch
  for (int j = 0; j < 8; ++j) {
    size_t rloc = row0 + rt * 16 + j + ro;
    kout[rloc * D_DIM + cb * 16 + ml]      = (__bf16)kacc0[j];
    kout[rloc * D_DIM + cb * 16 + 16 + ml] = (__bf16)kacc1[j];
    size_t nn = nb + rt * 16 + j + ro;
    vtout[(bl * D_DIM + cb * 16 + ml) * N_TOK + nn]      = (__bf16)vacc0[j];
    vtout[(bl * D_DIM + cb * 16 + 16 + ml) * N_TOK + nn] = (__bf16)vacc1[j];
  }
}

// ---------------------------------------------------------------------------
// slots_n = LN(slots); q = slots_n @ Wq^T (stored bf16 for async staging);
// zero upd/denom accumulators.  One block per batch of the group.
// ---------------------------------------------------------------------------
__global__ void __launch_bounds__(256) kn_slotpre(const float* __restrict__ slots,
                                                  const float* __restrict__ Wq,
                                                  float* __restrict__ slots_n,
                                                  __bf16* __restrict__ qout,
                                                  float* __restrict__ upd,
                                                  float* __restrict__ denom) {
  __shared__ float s_sn[S_SLOTS][D_DIM];
  const int b = blockIdx.x, t = threadIdx.x, w = t >> 5, L = t & 31;

  {  // LN: wave w -> slot w
    const float* sr = slots + ((size_t)b * S_SLOTS + w) * D_DIM;
    float s = 0.f, s2 = 0.f;
    float vv[4];
    for (int j = 0; j < 4; ++j) { vv[j] = sr[L + 32 * j]; s += vv[j]; s2 += vv[j] * vv[j]; }
    for (int m = 16; m >= 1; m >>= 1) { s += __shfl_xor(s, m); s2 += __shfl_xor(s2, m); }
    float mu = s * (1.f / 128.f);
    float rs = rsqrtf(s2 * (1.f / 128.f) - mu * mu + LN_EPS_);
    for (int j = 0; j < 4; ++j) {
      float nv = (vv[j] - mu) * rs;
      s_sn[w][L + 32 * j] = nv;
      slots_n[((size_t)b * S_SLOTS + w) * D_DIM + L + 32 * j] = nv;
    }
  }
  __syncthreads();

  for (int o = t; o < S_SLOTS * D_DIM; o += 256) {
    int s = o >> 7, d = o & 127;
    const float* wr = Wq + (size_t)d * D_DIM;
    float acc = 0.f;
    for (int i = 0; i < D_DIM; ++i) acc += s_sn[s][i] * wr[i];
    qout[(size_t)b * S_SLOTS * D_DIM + o] = (__bf16)acc;
  }
  for (int o = t; o < S_SLOTS * D_DIM; o += 256) upd[(size_t)b * S_SLOTS * D_DIM + o] = 0.f;
  if (t < S_SLOTS) denom[b * S_SLOTS + t] = 0.f;
}

// ---------------------------------------------------------------------------
// Fused logits (bf16 WMMA) + softmax over 8 slots + updates += attn^T v
// (bf16 WMMA, K=32).  q tile is DMA'd global->LDS with
// GLOBAL_LOAD_ASYNC_TO_LDS_B128 (ASYNCcnt) -- no VGPR roundtrip.
// Block = 256 threads (8 waves), one local batch, 512-row n-chunk.
// ---------------------------------------------------------------------------
__global__ void __launch_bounds__(256) kn_attn(const __bf16* __restrict__ kmat,
                                               const __bf16* __restrict__ vt,
                                               const __bf16* __restrict__ q,
                                               float* __restrict__ upd,
                                               float* __restrict__ denom) {
  __shared__ __bf16 s_q[16 * 136];        // q^T (bf16), slots padded to 16 w/ zeros
  __shared__ __bf16 s_at[8][16][48];      // per-wave attn^T tile [slot][n], padded
  const int t = threadIdx.x, w = t >> 5, L = t & 31;
  const int b = blockIdx.x >> 3, chunk = blockIdx.x & 3 ? blockIdx.x & 7 : blockIdx.x & 7;
  const int ml = L & 15;
  const int lo8 = (L < 16) ? 0 : 8;       // 16-bit frag K-run offset per lane half
  const int hi8 = (L < 16) ? 0 : 8;       // C/D frag row offset per lane half
  const bool sval = ml < S_SLOTS;

  // zero padded slot rows 8..15 (disjoint from async-written rows 0..7)
  for (int i = t; i < 8 * 136; i += 256) s_q[8 * 136 + i] = (__bf16)0.f;
  // async DMA q rows 0..7 (8x128 bf16 = 128 x 16B chunks) straight into LDS.
  // Generic address of a __shared__ object carries the wave-relative LDS
  // offset in its low 32 bits (ISA 10.2), which is what VDST wants.
  if (t < 128) {
    int s = t >> 4, d = (t & 15) * 8;
    const __bf16* src = q + (size_t)b * S_SLOTS * D_DIM + t * 8;
    unsigned dst = (unsigned)(size_t)(s_q + s * 136 + d);
    asm volatile("global_load_async_to_lds_b128 %0, %1, off"
                 :: "v"(dst), "v"(src) : "memory");
  }
  asm volatile("s_wait_asynccnt 0x0" ::: "memory");
  __syncthreads();

  const float scale = 0.08838834764831845f;  // 1/sqrt(128)
  float dacc = 0.f;
  v8f uacc[8];
  for (int c = 0; c < 8; ++c) uacc[c] = (v8f){0,0,0,0,0,0,0,0};

  for (int p = 0; p < 2; ++p) {
    const int n0 = chunk * 512 + p * 256 + w * 32;
    const __bf16* kb = kmat + ((size_t)b * N_TOK + n0) * D_DIM;
    const __bf16* vb = vt + (size_t)b * D_DIM * N_TOK + n0;

    if (p == 0) {  // prefetch next pass (global_prefetch_b8)
      __builtin_prefetch(kb + (size_t)256 * D_DIM + (size_t)ml * D_DIM, 0, 0);
      __builtin_prefetch(vb + 256 + (size_t)ml * N_TOK, 0, 0);
    }

    // ---- logits: two 16-row sets, K = 128 in 4 bf16 WMMAs each ----
    v8f lac[2];
    lac[0] = (v8f){0,0,0,0,0,0,0,0};
    lac[1] = (v8f){0,0,0,0,0,0,0,0};
    for (int k0 = 0; k0 < D_DIM; k0 += 32) {
      v16bf bq = cat8(*(const v8bf*)(s_q + ml * 136 + k0 + lo8),
                      *(const v8bf*)(s_q + ml * 136 + k0 + 16 + lo8));
      const __bf16* ka0 = kb + (size_t)ml * D_DIM + k0 + lo8;
      lac[0] = wmma_bf16(cat8(*(const v8bf*)(ka0), *(const v8bf*)(ka0 + 16)), bq, lac[0]);
      const __bf16* ka1 = kb + (size_t)(16 + ml) * D_DIM + k0 + lo8;
      lac[1] = wmma_bf16(cat8(*(const v8bf*)(ka1), *(const v8bf*)(ka1 + 16)), bq, lac[1]);
    }

    // ---- softmax over slots (cross-lane, groups of 8 lanes) ----
    for (int set = 0; set < 2; ++set) {
      for (int r = 0; r < 8; ++r) {
        float lv = lac[set][r] * scale;
        float m = sval ? lv : -1e30f;
        m = fmaxf(m, __shfl_xor(m, 1));
        m = fmaxf(m, __shfl_xor(m, 2));
        m = fmaxf(m, __shfl_xor(m, 4));
        float e = sval ? __expf(lv - m) : 0.f;
        float sm = e;
        sm += __shfl_xor(sm, 1); sm += __shfl_xor(sm, 2); sm += __shfl_xor(sm, 4);
        float at = sval ? (e / sm + EPS_) : 0.f;
        dacc += at;
        s_at[w][ml][set * 16 + r + hi8] = (__bf16)at;  // attn^T: [slot][n-local]
      }
    }

    // ---- updates += attn^T (16s x 32n) @ v (32n x 128d), K=32 ----
    v16bf af = cat8(*(const v8bf*)(&s_at[w][ml][lo8]),
                    *(const v8bf*)(&s_at[w][ml][16 + lo8]));
    for (int c = 0; c < 8; ++c) {
      const __bf16* vr = vb + (size_t)(c * 16 + ml) * N_TOK;
      v16bf bf_ = cat8(*(const v8bf*)(vr + lo8), *(const v8bf*)(vr + 16 + lo8));
      uacc[c] = wmma_bf16(af, bf_, uacc[c]);
    }
  }

  // ---- block partials -> global accumulators ----
  if (sval) atomicAdd(&denom[b * S_SLOTS + ml], dacc);
  if (L < 16) {  // D rows 0..7 = valid slots, rows 8..15 padding
    for (int c = 0; c < 8; ++c)
      for (int r = 0; r < 8; ++r)
        atomicAdd(&upd[((size_t)b * S_SLOTS + r) * D_DIM + c * 16 + ml], uacc[c][r]);
  }
}

// ---------------------------------------------------------------------------
// updates/denom -> GRU(updates, slots_n) -> +MLP(LN(.)) -> slots_out
// One block per batch of the group (~2.6 MFLOP/block, VALU).
// ---------------------------------------------------------------------------
__global__ void __launch_bounds__(256) kn_slotupd(const float* __restrict__ upd_raw,
                                                  const float* __restrict__ denom,
                                                  const float* __restrict__ slots_n,
                                                  const float* __restrict__ W_ih,
                                                  const float* __restrict__ W_hh,
                                                  const float* __restrict__ b_ih,
                                                  const float* __restrict__ b_hh,
                                                  const float* __restrict__ W1,
                                                  const float* __restrict__ b1,
                                                  const float* __restrict__ W2,
                                                  const float* __restrict__ b2,
                                                  float* __restrict__ slots_out) {
  __shared__ float s_x[S_SLOTS][D_DIM];
  __shared__ float s_h[S_SLOTS][D_DIM];
  __shared__ float s_hn[S_SLOTS][D_DIM];
  __shared__ float s_ln[S_SLOTS][D_DIM];
  __shared__ float s_hid[S_SLOTS][H_DIM];
  const int b = blockIdx.x, t = threadIdx.x, w = t >> 5, L = t & 31;

  for (int o = t; o < S_SLOTS * D_DIM; o += 256) {
    int s = o >> 7;
    s_x[s][o & 127] = upd_raw[(size_t)b * S_SLOTS * D_DIM + o] / denom[b * S_SLOTS + s];
    s_h[s][o & 127] = slots_n[(size_t)b * S_SLOTS * D_DIM + o];
  }
  __syncthreads();

  for (int o = t; o < S_SLOTS * D_DIM; o += 256) {
    int s = o >> 7, d = o & 127;
    const float* xv = s_x[s];
    const float* hv = s_h[s];
    float xr = b_ih[d], xz = b_ih[D_DIM + d], xn = b_ih[2 * D_DIM + d];
    float hr = b_hh[d], hz = b_hh[D_DIM + d], hn = b_hh[2 * D_DIM + d];
    const float* wir = W_ih + (size_t)d * D_DIM;
    const float* wiz = W_ih + (size_t)(D_DIM + d) * D_DIM;
    const float* win = W_ih + (size_t)(2 * D_DIM + d) * D_DIM;
    const float* whr = W_hh + (size_t)d * D_DIM;
    const float* whz = W_hh + (size_t)(D_DIM + d) * D_DIM;
    const float* whn = W_hh + (size_t)(2 * D_DIM + d) * D_DIM;
    for (int i = 0; i < D_DIM; ++i) {
      float xi = xv[i], hi = hv[i];
      xr += xi * wir[i]; xz += xi * wiz[i]; xn += xi * win[i];
      hr += hi * whr[i]; hz += hi * whz[i]; hn += hi * whn[i];
    }
    float r = 1.f / (1.f + __expf(-(xr + hr)));
    float z = 1.f / (1.f + __expf(-(xz + hz)));
    float n = tanhf(xn + r * hn);
    s_hn[s][d] = (1.f - z) * n + z * hv[d];
  }
  __syncthreads();

  {  // LN of new slots: wave w -> slot w
    float s = 0.f, s2 = 0.f;
    float vv[4];
    for (int j = 0; j < 4; ++j) { vv[j] = s_hn[w][L + 32 * j]; s += vv[j]; s2 += vv[j] * vv[j]; }
    for (int m = 16; m >= 1; m >>= 1) { s += __shfl_xor(s, m); s2 += __shfl_xor(s2, m); }
    float mu = s * (1.f / 128.f);
    float rs = rsqrtf(s2 * (1.f / 128.f) - mu * mu + LN_EPS_);
    for (int j = 0; j < 4; ++j) s_ln[w][L + 32 * j] = (vv[j] - mu) * rs;
  }
  __syncthreads();

  for (int o = t; o < S_SLOTS * H_DIM; o += 256) {
    int s = o >> 8, hh = o & 255;
    const float* wr = W1 + (size_t)hh * D_DIM;
    float acc = b1[hh];
    for (int i = 0; i < D_DIM; ++i) acc += s_ln[s][i] * wr[i];
    s_hid[s][hh] = fmaxf(acc, 0.f);
  }
  __syncthreads();

  for (int o = t; o < S_SLOTS * D_DIM; o += 256) {
    int s = o >> 7, d = o & 127;
    const float* wr = W2 + (size_t)d * H_DIM;
    float acc = b2[d];
    for (int i = 0; i < H_DIM; ++i) acc += s_hid[s][i] * wr[i];
    slots_out[(size_t)b * S_SLOTS * D_DIM + o] = s_hn[s][d] + acc;
  }
}

// ---------------------------------------------------------------------------
extern "C" void kernel_launch(void* const* d_in, const int* in_sizes, int n_in,
                              void* d_out, int out_size, void* d_ws, size_t ws_size,
                              hipStream_t stream) {
  (void)in_sizes; (void)n_in; (void)out_size; (void)ws_size;
  const float* inputs = (const float*)d_in[0];
  const float* noise  = (const float*)d_in[1];
  const float* smu    = (const float*)d_in[2];
  const float* sls    = (const float*)d_in[3];
  const float* Wq     = (const float*)d_in[4];
  const float* Wk     = (const float*)d_in[5];
  const float* Wv     = (const float*)d_in[6];
  const float* W_ih   = (const float*)d_in[7];
  const float* W_hh   = (const float*)d_in[8];
  const float* b_ih   = (const float*)d_in[9];
  const float* b_hh   = (const float*)d_in[10];
  const float* W1     = (const float*)d_in[11];
  const float* b1     = (const float*)d_in[12];
  const float* W2     = (const float*)d_in[13];
  const float* b2     = (const float*)d_in[14];

  const size_t KVG = (size_t)GROUPB * N_TOK * D_DIM;     // bf16 elems per group buffer
  const size_t SL  = (size_t)B_BATCH * S_SLOTS * D_DIM;  // 131,072
  const size_t SLB = (size_t)S_SLOTS * D_DIM;            // 1024 per batch

  __bf16* kbf  = (__bf16*)d_ws;        // group k, reused by every group (stays in L2)
  __bf16* vtb  = kbf + KVG;            // group v^T, reused by every group
  __bf16* qbuf = vtb + KVG;            // q in bf16 (async-staged by kn_attn)
  float*  fws  = (float*)(qbuf + SL);
  float* slots  = fws;  fws += SL;
  float* slotsn = fws;  fws += SL;
  float* upd    = fws;  fws += SL;
  float* den    = fws;  fws += (size_t)B_BATCH * S_SLOTS;

  kn_init<<<(int)((SL + 255) / 256), 256, 0, stream>>>(noise, smu, sls, slots);

  for (int g = 0; g < B_BATCH / GROUPB; ++g) {
    const size_t b0 = (size_t)g * GROUPB;
    const float* xg = inputs + b0 * N_TOK * D_DIM;

    kn_lnkv<<<(GROUPB * N_TOK) / 32, 256, 0, stream>>>(xg, Wk, Wv, kbf, vtb);

    for (int it = 0; it < NUM_ITER_; ++it) {
      kn_slotpre<<<GROUPB, 256, 0, stream>>>(slots + b0 * SLB, Wq, slotsn + b0 * SLB,
                                             qbuf + b0 * SLB, upd + b0 * SLB,
                                             den + b0 * S_SLOTS);
      kn_attn<<<GROUPB * 8, 256, 0, stream>>>(kbf, vtb, qbuf + b0 * SLB,
                                              upd + b0 * SLB, den + b0 * S_SLOTS);
      float* outp = (it == NUM_ITER_ - 1) ? ((float*)d_out + b0 * SLB)
                                          : (slots + b0 * SLB);
      kn_slotupd<<<GROUPB, 256, 0, stream>>>(upd + b0 * SLB, den + b0 * S_SLOTS,
                                             slotsn + b0 * SLB, W_ih, W_hh, b_ih, b_hh,
                                             W1, b1, W2, b2, outp);
    }
  }
}
